// Module_22016002359310
// MI455X (gfx1250) — matrix operand order repaired and checked by reading, not compiled, not run on hardware
//
#include <hip/hip_runtime.h>

typedef _Float16 v16h __attribute__((ext_vector_type(16)));
typedef _Float16 v8h  __attribute__((ext_vector_type(8)));
typedef _Float16 v4h  __attribute__((ext_vector_type(4)));
typedef float    v8f  __attribute__((ext_vector_type(8)));
typedef float    v4f  __attribute__((ext_vector_type(4)));

#define WIDTH    64
#define RW       32       // batch rows per wave (2 M-tiles -> B-fragment reuse)
#define BT       256      // batch rows per workgroup (8 waves x 32 rows)
#define AST      72       // padded LDS activation row stride in f16 (144B)
#define NLAYERS  5

// ---------------------------------------------------------------------------
// Prep: convert the five [o][i] fp32 weight matrices to f16, kept as
// [n=o][k=i] row-major, packed as wt[5][64][64] in workspace. Each WMMA
// B-fragment is then two 16B loads per lane (lane = N column, 16 K values).
// ---------------------------------------------------------------------------
__global__ void prep_weights_kernel(const float* __restrict__ w_in,
                                    const float* __restrict__ w_h,
                                    const float* __restrict__ w_out,
                                    _Float16* __restrict__ wt) {
    int e = blockIdx.x * 256 + threadIdx.x;          // 0 .. 5*4096-1
    if (e >= NLAYERS * WIDTH * WIDTH) return;
    int l = e >> 12;          // layer
    int r = e & 4095;
    int o = r >> 6;           // N (output feature)
    int i = r & 63;           // K (input feature)
    float v;
    if (l == 0)      v = w_in [o * WIDTH + i];
    else if (l <= 3) v = w_h  [(l - 1) * WIDTH * WIDTH + o * WIDTH + i];
    else             v = w_out[o * WIDTH + i];
    wt[e] = (_Float16)v;
}

// ---------------------------------------------------------------------------
// Fully fused MLP. Each wave owns 32 batch rows (two 16-row M-tiles) for the
// whole network, so every B-fragment global load feeds TWO WMMAs:
//   - stage 32 rows of x (fp32 -> f16, non-temporal) into a private LDS slice
//   - per layer: A-fragments (ISA 16x32 f16 layout) from LDS, B-fragments
//     from L0/L2-resident f16 [n][k] weights, v_wmma_f32_16x16x32_f16 with
//     f32 accumulate (K=64 = 2 steps), ReLU + f16 back to LDS
//   - last layer: f32 accumulators stored non-temporally to HBM.
// No cross-wave sharing -> no barriers; DS ops are in-order per wave.
// ---------------------------------------------------------------------------
__global__ __launch_bounds__(256) void mlp_fused_kernel(
        const float* __restrict__ x,
        const _Float16* __restrict__ wt,
        float* __restrict__ out) {
    __shared__ _Float16 act[BT * AST];               // 36,864 B

    const int tid  = threadIdx.x;
    const int wave = tid >> 5;
    const int lane = tid & 31;
    const int m    = lane & 15;                      // A row / C column index
    const int half = lane >> 4;                      // lane half select

    const size_t rowWave = (size_t)blockIdx.x * BT + (size_t)wave * RW;

    // ---- stage this wave's 32 input rows: fp32 global -> f16 LDS ----
    {
        const v4f* xs = (const v4f*)(x + rowWave * WIDTH);   // 16B-aligned
        _Float16* as = &act[(wave * RW) * AST];
        #pragma unroll
        for (int i = 0; i < 16; ++i) {
            int idx = lane + i * 32;                 // 512 float4 chunks / wave
            int row = idx >> 4;
            int c4  = idx & 15;
            v4f v = __builtin_nontemporal_load(&xs[row * 16 + c4]);
            v4h h;
            h.x = (_Float16)v.x; h.y = (_Float16)v.y;
            h.z = (_Float16)v.z; h.w = (_Float16)v.w;
            *(v4h*)&as[row * AST + c4 * 4] = h;      // ds_store_b64
        }
    }

    // A rows this lane reads for the two M-tiles (lane m holds row m)
    _Float16* arow0 = &act[(wave * RW +      m) * AST];
    _Float16* arow1 = &act[(wave * RW + 16 + m) * AST];

    #pragma unroll
    for (int l = 0; l < NLAYERS; ++l) {
        const _Float16* wl = wt + l * (WIDTH * WIDTH);

        // A fragments for both M-tiles, K = 0..63 in two K=32 steps.
        // v16h halves: [0..7]=A[m][s*32+half*8..], [8..15]=A[m][s*32+16+half*8..]
        v8h p0, p1;
        p0 = *(const v8h*)(arow0 +  0 + half * 8);
        p1 = *(const v8h*)(arow0 + 16 + half * 8);
        v16h a00 = __builtin_shufflevector(p0, p1, 0,1,2,3,4,5,6,7,8,9,10,11,12,13,14,15);
        p0 = *(const v8h*)(arow0 + 32 + half * 8);
        p1 = *(const v8h*)(arow0 + 48 + half * 8);
        v16h a01 = __builtin_shufflevector(p0, p1, 0,1,2,3,4,5,6,7,8,9,10,11,12,13,14,15);
        p0 = *(const v8h*)(arow1 +  0 + half * 8);
        p1 = *(const v8h*)(arow1 + 16 + half * 8);
        v16h a10 = __builtin_shufflevector(p0, p1, 0,1,2,3,4,5,6,7,8,9,10,11,12,13,14,15);
        p0 = *(const v8h*)(arow1 + 32 + half * 8);
        p1 = *(const v8h*)(arow1 + 48 + half * 8);
        v16h a11 = __builtin_shufflevector(p0, p1, 0,1,2,3,4,5,6,7,8,9,10,11,12,13,14,15);

        #pragma unroll
        for (int nt = 0; nt < 4; ++nt) {
            // B fragments: lane = N column m, halves = K in same order as A.
            // Each is reused by BOTH M-tiles (2 WMMAs per load pair).
            const _Float16* bcol = wl + (nt * 16 + m) * WIDTH + half * 8;
            v8h q0 = *(const v8h*)(bcol +  0);
            v8h q1 = *(const v8h*)(bcol + 16);
            v16h b0 = __builtin_shufflevector(q0, q1, 0,1,2,3,4,5,6,7,8,9,10,11,12,13,14,15);
            q0 = *(const v8h*)(bcol + 32);
            q1 = *(const v8h*)(bcol + 48);
            v16h b1 = __builtin_shufflevector(q0, q1, 0,1,2,3,4,5,6,7,8,9,10,11,12,13,14,15);

            v8f c0 = {};
            v8f c1 = {};
            c0 = __builtin_amdgcn_wmma_f32_16x16x32_f16(false, a00, false, b0,
                                                        (short)0, c0, false, false);
            c1 = __builtin_amdgcn_wmma_f32_16x16x32_f16(false, a10, false, b0,
                                                        (short)0, c1, false, false);
            c0 = __builtin_amdgcn_wmma_f32_16x16x32_f16(false, a01, false, b1,
                                                        (short)0, c0, false, false);
            c1 = __builtin_amdgcn_wmma_f32_16x16x32_f16(false, a11, false, b1,
                                                        (short)0, c1, false, false);

            if (l < NLAYERS - 1) {
                // ReLU, f32->f16, scatter C-layout back to LDS rows
                _Float16* d0 = &act[(wave * RW +      8 * half) * AST + nt * 16 + m];
                _Float16* d1 = &act[(wave * RW + 16 + 8 * half) * AST + nt * 16 + m];
                #pragma unroll
                for (int r = 0; r < 8; ++r) {
                    float v0 = fmaxf(c0[r], 0.0f);
                    float v1 = fmaxf(c1[r], 0.0f);
                    d0[r * AST] = (_Float16)v0;
                    d1[r * AST] = (_Float16)v1;
                }
            } else {
                // linear output layer: fp32 straight to HBM, non-temporal
                float* o0 = out + (rowWave +      8 * half) * WIDTH + nt * 16 + m;
                float* o1 = out + (rowWave + 16 + 8 * half) * WIDTH + nt * 16 + m;
                #pragma unroll
                for (int r = 0; r < 8; ++r) {
                    __builtin_nontemporal_store(c0[r], o0 + (size_t)r * WIDTH);
                    __builtin_nontemporal_store(c1[r], o1 + (size_t)r * WIDTH);
                }
            }
        }
    }
}

extern "C" void kernel_launch(void* const* d_in, const int* in_sizes, int n_in,
                              void* d_out, int out_size, void* d_ws, size_t ws_size,
                              hipStream_t stream) {
    const float* x     = (const float*)d_in[0];
    const float* w_in  = (const float*)d_in[1];
    const float* w_h   = (const float*)d_in[2];
    const float* w_out = (const float*)d_in[3];
    float* out = (float*)d_out;
    _Float16* wt = (_Float16*)d_ws;                  // 40 KB of workspace

    const int nrows = in_sizes[0] / WIDTH;           // 1,048,576

    prep_weights_kernel<<<(NLAYERS * WIDTH * WIDTH + 255) / 256, 256, 0, stream>>>(
        w_in, w_h, w_out, wt);
    mlp_fused_kernel<<<nrows / BT, 256, 0, stream>>>(x, wt, out);
}
